// GCNBlock_14620068675981
// MI455X (gfx1250) — matrix operand order repaired and checked by reading, not compiled, not run on hardware
//
#include <hip/hip_runtime.h>

// GCN block: y = LN_global(mask)( (adj @ x) @ W + bias ) * mask, relu.
// B=64, N=512, DIN=DOUT=256.
// fp32 in HBM -> v_cvt_pk_bf16_f32 during LDS staging ->
// v_wmma_f32_16x16x32_bf16 (fp32 accumulate). LDS double-buffered;
// global_prefetch_b8 two K-steps ahead.

typedef __attribute__((ext_vector_type(16))) __bf16 v16bf;
typedef __attribute__((ext_vector_type(8)))  float  v8f;

union FragAB { v16bf bf; uint4 q[2]; };

#define TM 128          // block tile rows (M)
#define TN 128          // block tile cols (N)
#define TK 32           // K per WMMA step (bf16)
#define LS 40           // LDS row stride in bf16: 32 + 8 pad (80 B, 16B-aligned)

// ---- fp32x2 -> packed bf16x2 ---------------------------------------------
#if defined(__has_builtin)
#if __has_builtin(__builtin_amdgcn_cvt_pk_bf16_f32)
#define HAVE_CVT_PK_BF16 1
#endif
#endif

__device__ __forceinline__ unsigned cvt_pk_bf16(float a, float b) {
#ifdef HAVE_CVT_PK_BF16
    typedef __attribute__((ext_vector_type(2))) __bf16 v2bf;
    union { v2bf v; unsigned u; } r;
    r.v = __builtin_amdgcn_cvt_pk_bf16_f32(a, b);
    return r.u;
#else
    // Hardware packed convert: a -> [15:0], b -> [31:16]
    unsigned r;
    asm("v_cvt_pk_bf16_f32 %0, %1, %2" : "=v"(r) : "v"(a), "v"(b));
    return r;
#endif
}
// --------------------------------------------------------------------------

// Fetch one K-step's A (128x32) and B (32x128) tiles into registers (fp32).
__device__ __forceinline__ void fetch_tiles(const float* __restrict__ Ab,
                                            const float* __restrict__ Bb,
                                            int m0, int n0, int k0,
                                            int lda, int ldb, int tid,
                                            float4 ra[4], float4 rb[4]) {
    #pragma unroll
    for (int i = 0; i < 4; ++i) {                 // A: 1024 float4, 4/thread
        int e = tid + i * 256;
        int row = e >> 3, c4 = e & 7;
        ra[i] = *reinterpret_cast<const float4*>(
            Ab + (long long)(m0 + row) * lda + k0 + c4 * 4);
    }
    #pragma unroll
    for (int i = 0; i < 4; ++i) {                 // B: 1024 float4, 4/thread
        int e = tid + i * 256;
        int kr = e >> 5, c4 = e & 31;
        rb[i] = *reinterpret_cast<const float4*>(
            Bb + (long long)(k0 + kr) * ldb + n0 + c4 * 4);
    }
}

// Convert fp32 regs -> bf16 and commit to an LDS buffer (A row-major, B transposed).
__device__ __forceinline__ void commit_tiles(unsigned short* __restrict__ dA,
                                             unsigned short* __restrict__ dB,
                                             int tid,
                                             const float4 ra[4], const float4 rb[4]) {
    #pragma unroll
    for (int i = 0; i < 4; ++i) {
        int e = tid + i * 256;
        int row = e >> 3, c4 = e & 7;
        // packed pair stores: 8B-aligned (80*row + 8*c4) -> ds_store_b64
        uint2 p;
        p.x = cvt_pk_bf16(ra[i].x, ra[i].y);
        p.y = cvt_pk_bf16(ra[i].z, ra[i].w);
        *reinterpret_cast<uint2*>(&dA[row * LS + c4 * 4]) = p;
    }
    #pragma unroll
    for (int i = 0; i < 4; ++i) {
        int e = tid + i * 256;
        int kr = e >> 5, c4 = e & 31;
        int n = c4 * 4;
        unsigned p0 = cvt_pk_bf16(rb[i].x, rb[i].y);
        unsigned p1 = cvt_pk_bf16(rb[i].z, rb[i].w);
        // low halves -> ds_store_b16, high halves -> ds_store_b16_d16_hi
        dB[(n + 0) * LS + kr] = (unsigned short)p0;
        dB[(n + 1) * LS + kr] = (unsigned short)(p0 >> 16);
        dB[(n + 2) * LS + kr] = (unsigned short)p1;
        dB[(n + 3) * LS + kr] = (unsigned short)(p1 >> 16);
    }
}

// C[b] = A[b](MxK) * Bm[b](KxN) (+bias). M%128==0, N%128==0, K%32==0.
__global__ __launch_bounds__(256)
void gemm_bf16_wmma(const float* __restrict__ A, const float* __restrict__ Bm,
                    float* __restrict__ C, const float* __restrict__ bias,
                    int K, int lda, int ldb, int ldc,
                    long long sAb, long long sBb, long long sCb)
{
    __shared__ __align__(16) unsigned short sA[2][TM * LS];   // [m][k] bf16
    __shared__ __align__(16) unsigned short sB[2][TN * LS];   // [n][k] bf16 (transposed)

    const int tid  = threadIdx.x;
    const int wid  = tid >> 5;        // 8 waves -> 16-row stripes of M
    const int lane = tid & 31;
    const int half = lane >> 4;
    const int l16  = lane & 15;

    const int m0 = blockIdx.y * TM;
    const int n0 = blockIdx.x * TN;
    const int bz = blockIdx.z;

    const float* Ab = A  + (long long)bz * sAb;
    const float* Bb = Bm + (long long)bz * sBb;
    float*       Cb = C  + (long long)bz * sCb;

    v8f acc[8];
    #pragma unroll
    for (int j = 0; j < 8; ++j) acc[j] = {};

    const int nsteps = K / TK;

    // Prologue: stage step 0 into buffer 0.
    {
        float4 ra[4], rb[4];
        fetch_tiles(Ab, Bb, m0, n0, 0, lda, ldb, tid, ra, rb);
        commit_tiles(sA[0], sB[0], tid, ra, rb);
    }

    for (int s = 0; s < nsteps; ++s) {
        __syncthreads();
        const int cur = s & 1;
        const bool hasNext = (s + 1) < nsteps;

        // (a) Issue next step's global loads first (latency hidden by WMMA burst).
        float4 ra[4], rb[4];
        if (hasNext)
            fetch_tiles(Ab, Bb, m0, n0, (s + 1) * TK, lda, ldb, tid, ra, rb);

        // L2 prefetch two K-steps ahead (global_prefetch_b8).
        if (s + 2 < nsteps) {
            const int kp = (s + 2) * TK;
            if (tid < 128) {
                // A: one 128B line per row (32 fp32)
                __builtin_prefetch(Ab + (long long)(m0 + tid) * lda + kp, 0, 0);
            } else {
                // B: 32 rows x 512B slice -> 4 lines per row
                int idx = tid - 128;
                int kr = idx >> 2, seg = (idx & 3) * 32;
                __builtin_prefetch(Bb + (long long)(kp + kr) * ldb + n0 + seg, 0, 0);
            }
        }

        // (b) Compute from the current buffer.
        // A fragment (16x32 bf16): V0-7 = K half*16..+15 (contiguous, matches B).
        FragAB a;
        const int ml = wid * 16 + l16;
        a.q[0] = *reinterpret_cast<const uint4*>(&sA[cur][ml * LS + half * 16]);
        a.q[1] = *reinterpret_cast<const uint4*>(&sA[cur][ml * LS + half * 16 + 8]);

        #pragma unroll
        for (int g = 0; g < 2; ++g) {
            FragAB bf[4];
            #pragma unroll
            for (int j = 0; j < 4; ++j) {
                const int nl = (g * 4 + j) * 16 + l16;
                bf[j].q[0] = *reinterpret_cast<const uint4*>(&sB[cur][nl * LS + half * 16]);
                bf[j].q[1] = *reinterpret_cast<const uint4*>(&sB[cur][nl * LS + half * 16 + 8]);
            }
            #pragma unroll
            for (int j = 0; j < 4; ++j) {
                acc[g * 4 + j] = __builtin_amdgcn_wmma_f32_16x16x32_bf16(
                    false, a.bf, false, bf[j].bf, (short)0, acc[g * 4 + j], false, false);
            }
        }

        // (c) Commit next tile into the other buffer (protected by next barrier).
        if (hasNext)
            commit_tiles(sA[cur ^ 1], sB[cur ^ 1], tid, ra, rb);
    }

    // Epilogue: D layout lane = n (0..15), VGPR r -> m = r + 8*half.
    #pragma unroll
    for (int j = 0; j < 8; ++j) {
        int ng = n0 + j * 16 + l16;
        float bv = bias ? bias[ng] : 0.0f;
        #pragma unroll
        for (int r = 0; r < 8; ++r) {
            int mg = m0 + wid * 16 + r + 8 * half;
            Cb[(long long)mg * ldc + ng] = acc[j][r] + bv;
        }
    }
}

__global__ void zero_stats(float* __restrict__ stats, int n) {
    int i = blockIdx.x * blockDim.x + threadIdx.x;
    if (i < n) stats[i] = 0.0f;
}

// Per-channel masked sum / sumsq / count. thread = channel (D=256).
__global__ __launch_bounds__(256)
void reduce_stats(const float* __restrict__ y, const float* __restrict__ mask,
                  float* __restrict__ stats, int rowsPerBlock, int D)
{
    const int d  = threadIdx.x;
    const int r0 = blockIdx.x * rowsPerBlock;
    float s = 0.0f, s2 = 0.0f, c = 0.0f;
    for (int r = r0; r < r0 + rowsPerBlock; ++r) {
        float m = mask[r];
        float v = y[(long long)r * D + d];
        s  += m * v;
        s2 += m * v * v;
        if (d == 0) c += m;
    }
    atomicAdd(&stats[d], s);
    atomicAdd(&stats[D + d], s2);
    if (d == 0) atomicAdd(&stats[2 * D], c);
}

// out = relu(((y-mean)*rsqrt(var+eps)*gamma + beta) * mask)
__global__ __launch_bounds__(256)
void normalize_kernel(const float* __restrict__ y, const float* __restrict__ mask,
                      const float* __restrict__ gamma, const float* __restrict__ beta,
                      const float* __restrict__ stats, float* __restrict__ out,
                      long long total, int D)
{
    long long i = (long long)blockIdx.x * blockDim.x + threadIdx.x;
    if (i >= total) return;
    int d       = (int)(i & (D - 1));
    long long r = i >> 8;                 // D == 256
    float cnt  = stats[2 * D];
    float mean = stats[d] / cnt;
    float var  = stats[D + d] / cnt - mean * mean;
    float rs   = rsqrtf(var + 1e-5f);
    float z    = ((y[i] - mean) * rs * gamma[d] + beta[d]) * mask[r];
    out[i] = fmaxf(z, 0.0f);
}

extern "C" void kernel_launch(void* const* d_in, const int* in_sizes, int n_in,
                              void* d_out, int out_size, void* d_ws, size_t ws_size,
                              hipStream_t stream) {
    (void)in_sizes; (void)n_in; (void)out_size; (void)ws_size;
    const float* x      = (const float*)d_in[0];   // (B,N,DIN)
    const float* adj    = (const float*)d_in[1];   // (B,N,N)
    const float* mask   = (const float*)d_in[2];   // (B,N)
    const float* weight = (const float*)d_in[3];   // (DIN,DOUT)
    const float* bias   = (const float*)d_in[4];   // (DOUT)
    const float* gamma  = (const float*)d_in[5];   // (DOUT)
    const float* beta   = (const float*)d_in[6];   // (DOUT)
    float* out = (float*)d_out;

    const int B = 64, N = 512, DIN = 256, DOUT = 256;

    float* t     = (float*)d_ws;                    // B*N*DIN   = 33.5 MB
    float* y     = t + (size_t)B * N * DIN;         // B*N*DOUT  = 33.5 MB
    float* stats = y + (size_t)B * N * DOUT;        // 2*DOUT+1 floats

    // K1: t[b] = adj[b] (512x512) @ x[b] (512x256)
    dim3 g1(DIN / TN, N / TM, B);
    gemm_bf16_wmma<<<g1, 256, 0, stream>>>(
        adj, x, t, nullptr,
        /*K=*/N, /*lda=*/N, /*ldb=*/DIN, /*ldc=*/DIN,
        (long long)N * N, (long long)N * DIN, (long long)N * DIN);

    // K2: y = t (32768x256) @ weight (256x256) + bias
    dim3 g2(DOUT / TN, (B * N) / TM, 1);
    gemm_bf16_wmma<<<g2, 256, 0, stream>>>(
        t, weight, y, bias,
        /*K=*/DIN, /*lda=*/DIN, /*ldb=*/DOUT, /*ldc=*/DOUT,
        0LL, 0LL, 0LL);

    // K3: stats
    zero_stats<<<(2 * DOUT + 1 + 255) / 256, 256, 0, stream>>>(stats, 2 * DOUT + 1);
    reduce_stats<<<64, 256, 0, stream>>>(y, mask, stats, (B * N) / 64, DOUT);

    // K4: normalize + mask + relu
    long long total = (long long)B * N * DOUT;
    normalize_kernel<<<(unsigned)((total + 255) / 256), 256, 0, stream>>>(
        y, mask, gamma, beta, stats, out, total, DOUT);
}
